// LlamaAttention_84688165143128
// MI455X (gfx1250) — matrix operand order repaired and checked by reading, not compiled, not run on hardware
//
#include <hip/hip_runtime.h>

// ---------------- types ----------------
typedef __attribute__((ext_vector_type(16))) __bf16        v16bf;
typedef __attribute__((ext_vector_type(8)))  float          v8f;
typedef __attribute__((ext_vector_type(8)))  unsigned short us8;
typedef __attribute__((ext_vector_type(16))) unsigned short us16;
typedef __attribute__((ext_vector_type(4)))  float          f4;

#define DEV __device__ __forceinline__

// ---------------- helpers ----------------
DEV unsigned short f2bf(float f) {
  unsigned int u = __builtin_bit_cast(unsigned int, f);
  u += 0x7fffu + ((u >> 16) & 1u);           // round-to-nearest-even
  return (unsigned short)(u >> 16);
}
DEV float bf2f(unsigned short h) {
  unsigned int u = ((unsigned int)h) << 16;
  return __builtin_bit_cast(float, u);
}
DEV v8f vzero8() {
  v8f z;
#pragma unroll
  for (int i = 0; i < 8; ++i) z[i] = 0.0f;
  return z;
}
DEV v16bf mkfrag(us8 lo, us8 hi) {
  us16 t;
#pragma unroll
  for (int i = 0; i < 8; ++i) { t[i] = lo[i]; t[i + 8] = hi[i]; }
  return __builtin_bit_cast(v16bf, t);
}
// B-fragment: two 8-elem runs at p and p+16 (same k order as A)
DEV v16bf ldfrag(const unsigned short* p) {
  us8 lo = *(const us8*)(p);
  us8 hi = *(const us8*)(p + 16);
  return mkfrag(lo, hi);
}
// A-fragment: two 8-elem runs at p and p+16 (per ISA 16-bit A layout)
DEV v16bf ldfragA(const unsigned short* p) {
  us8 lo = *(const us8*)(p);
  us8 hi = *(const us8*)(p + 16);
  return mkfrag(lo, hi);
}
DEV v8f wmma_bf16(v16bf a, v16bf b, v8f c) {
  return __builtin_amdgcn_wmma_f32_16x16x32_bf16(false, a, false, b, (short)0, c,
                                                 false, false);
}
// Async copy 32B (two b128) global->LDS; offset applies to both addresses.
DEV void async_copy32(unsigned lds_off, const unsigned short* gsrc) {
  asm volatile(
      "global_load_async_to_lds_b128 %0, %1, off\n\t"
      "global_load_async_to_lds_b128 %0, %1, off offset:16"
      :: "v"(lds_off), "v"(gsrc) : "memory");
}
DEV void async_copy64(unsigned lds_off, const unsigned short* gsrc) {
  asm volatile(
      "global_load_async_to_lds_b128 %0, %1, off\n\t"
      "global_load_async_to_lds_b128 %0, %1, off offset:16\n\t"
      "global_load_async_to_lds_b128 %0, %1, off offset:32\n\t"
      "global_load_async_to_lds_b128 %0, %1, off offset:48"
      :: "v"(lds_off), "v"(gsrc) : "memory");
}
DEV void wait_async0() { asm volatile("s_wait_asynccnt 0x0" ::: "memory"); }
DEV unsigned lds_off(const void* p) { return (unsigned)(size_t)p; }

// ---------------- f32 -> bf16 ----------------
__global__ __launch_bounds__(256) void cvt_f32_bf16(const float* __restrict__ src,
                                                    unsigned short* __restrict__ dst,
                                                    int n) {
  int i = blockIdx.x * 256 + threadIdx.x;
  if (i < n) dst[i] = f2bf(src[i]);
}

// ---------------- NT GEMM: C[M,N] = sum_K A[M,K] * W[N,K]  (bf16 in, f32 acc)
// Block tile 128x128, 8 waves of 32Mx64N, K step 32.
// Double-buffered LDS staging via GLOBAL_LOAD_ASYNC_TO_LDS_B128 (ASYNCcnt).
template <bool OUTBF16, bool HEADL>
__global__ __launch_bounds__(256) void gemm_nt(const unsigned short* __restrict__ A,
                                               const unsigned short* __restrict__ W,
                                               void* __restrict__ Cp,
                                               int M, int N, int K) {
  __shared__ unsigned short Asm[2][128 * 40];  // 128 rows x 32 k (+8 pad)
  __shared__ unsigned short Wsm[2][128 * 40];  // 128 cols x 32 k (+8 pad)

  const int t = threadIdx.x;
  const int lane = t & 31;
  const int w = t >> 5;
  const int rsel = lane & 15;
  const int hsel = lane >> 4;
  const int m0 = blockIdx.y * 128;
  const int n0 = blockIdx.x * 128;
  const int wm = (w & 3) * 32;
  const int wn = (w >> 2) * 64;

  v8f acc[2][4];
#pragma unroll
  for (int i = 0; i < 2; ++i)
#pragma unroll
    for (int j = 0; j < 4; ++j) acc[i][j] = vzero8();

  const int lrow = t >> 1;          // 0..127
  const int lk = (t & 1) * 16;      // 0 or 16
  const unsigned short* ag = A + (size_t)(m0 + lrow) * K + lk;
  const unsigned short* wg = W + (size_t)(n0 + lrow) * K + lk;

  const int nk = K / 32;
  // prologue: stage tile 0
  async_copy32(lds_off(&Asm[0][lrow * 40 + lk]), ag);
  async_copy32(lds_off(&Wsm[0][lrow * 40 + lk]), wg);
  wait_async0();
  __syncthreads();

  for (int ki = 0; ki < nk; ++ki) {
    const int buf = ki & 1;
    if (ki + 1 < nk) {  // stage next tile while computing (uniform branch)
      const int nb = (ki + 1) & 1;
      async_copy32(lds_off(&Asm[nb][lrow * 40 + lk]), ag + (ki + 1) * 32);
      async_copy32(lds_off(&Wsm[nb][lrow * 40 + lk]), wg + (ki + 1) * 32);
    }

    v16bf af[2], bf_[4];
#pragma unroll
    for (int mi = 0; mi < 2; ++mi)
      af[mi] = ldfragA(&Asm[buf][(wm + mi * 16 + rsel) * 40 + hsel * 8]);
#pragma unroll
    for (int ni = 0; ni < 4; ++ni)
      bf_[ni] = ldfrag(&Wsm[buf][(wn + ni * 16 + rsel) * 40 + hsel * 8]);
#pragma unroll
    for (int mi = 0; mi < 2; ++mi)
#pragma unroll
      for (int ni = 0; ni < 4; ++ni)
        acc[mi][ni] = wmma_bf16(af[mi], bf_[ni], acc[mi][ni]);

    wait_async0();     // next tile landed
    __syncthreads();   // my reads of buf are done before anyone overwrites it
  }

  // ---- store C tile ----
#pragma unroll
  for (int mi = 0; mi < 2; ++mi)
#pragma unroll
    for (int ni = 0; ni < 4; ++ni)
#pragma unroll
      for (int r = 0; r < 8; ++r) {
        const int row = m0 + wm + mi * 16 + r + 8 * hsel;
        const int col = n0 + wn + ni * 16 + rsel;
        const float v = acc[mi][ni][r];
        if (OUTBF16) {
          size_t idx;
          if (HEADL) {  // [B,H,S,hd] with S=2048, H=16, hd=128
            const int b = row >> 11, s = row & 2047;
            const int h = col >> 7, dd = col & 127;
            idx = ((((size_t)b * 16 + h) * 2048) + s) * 128 + dd;
          } else {
            idx = (size_t)row * N + col;
          }
          ((unsigned short*)Cp)[idx] = f2bf(v);
        } else {
          ((float*)Cp)[(size_t)row * N + col] = v;
        }
      }
}

// ---------------- RoPE in place on [B,H,S,hd] bf16 ----------------
__global__ __launch_bounds__(256) void rope_inplace(unsigned short* __restrict__ th,
                                                    const float* __restrict__ fc,
                                                    const float* __restrict__ fs) {
  const int i = blockIdx.x * 256 + threadIdx.x;  // one (t0,t1) pair
  const int fi = i & 63;
  const int s = (i >> 6) & 2047;
  const int bh = i >> 17;
  const size_t base = ((((size_t)bh * 2048) + s) << 7) + fi * 2;
  const float t0 = bf2f(th[base]);
  const float t1 = bf2f(th[base + 1]);
  const float c = fc[s * 64 + fi];
  const float sn = fs[s * 64 + fi];
  th[base] = f2bf(t0 * c - t1 * sn);
  th[base + 1] = f2bf(t0 * sn + t1 * c);
}

// ---------------- causal flash attention ----------------
// Grid (S/128, H, B); 8 waves x 16 query rows; 64-key LDS tiles.
__global__ __launch_bounds__(256) void flash_attn(const unsigned short* __restrict__ Q,
                                                  const unsigned short* __restrict__ Kk,
                                                  const unsigned short* __restrict__ V,
                                                  unsigned short* __restrict__ O) {
  __shared__ unsigned short Ksm[64 * 136];     // [key][d]  (pad 136)
  __shared__ unsigned short Vts[128 * 72];     // [d][key]  (pad 72)
  __shared__ unsigned short Psm[8 * 16 * 72];  // per-wave P tile 16x64

  const int t = threadIdx.x;
  const int lane = t & 31;
  const int w = t >> 5;
  const int rsel = lane & 15;
  const int hsel = lane >> 4;
  const int qtile = blockIdx.x;
  const int h = blockIdx.y;
  const int b = blockIdx.z;
  const int bh = b * 16 + h;
  const int q0 = qtile * 128 + w * 16;
  const size_t headoff = (size_t)bh * 2048 * 128;
  const float scale = 0.08838834764831845f;  // 1/sqrt(128)

  // Q fragments for this wave's 16 rows (held for the whole pass)
  const unsigned short* qrow = Q + headoff + (size_t)(q0 + rsel) * 128;
  v16bf qf[4];
#pragma unroll
  for (int f = 0; f < 4; ++f) qf[f] = ldfragA(qrow + f * 32 + hsel * 8);

  v8f oacc[8];
#pragma unroll
  for (int i = 0; i < 8; ++i) oacc[i] = vzero8();
  float m_s[8], l_s[8];
#pragma unroll
  for (int r = 0; r < 8; ++r) { m_s[r] = -1e30f; l_s[r] = 0.0f; }

  const int ntiles = 2 * qtile + 2;  // causal bound (uniform per block)
  for (int kt = 0; kt < ntiles; ++kt) {
    const int kbase = kt * 64;
    __syncthreads();  // previous tile fully consumed
    {
      const int row = t >> 2;       // 0..63 key in tile
      const int c0 = (t & 3) * 32;  // 0/32/64/96 dim
      // K tile: async copy straight to LDS (ASYNCcnt)
      async_copy64(lds_off(&Ksm[row * 136 + c0]),
                   Kk + headoff + (size_t)(kbase + row) * 128 + c0);
      // V tile: load + transpose scatter into [d][key]
      const unsigned short* vs = V + headoff + (size_t)(kbase + row) * 128 + c0;
      us8 b0 = *(const us8*)(vs), b1 = *(const us8*)(vs + 8);
      us8 b2 = *(const us8*)(vs + 16), b3 = *(const us8*)(vs + 24);
#pragma unroll
      for (int i = 0; i < 8; ++i) {
        Vts[(c0 + i) * 72 + row] = b0[i];
        Vts[(c0 + 8 + i) * 72 + row] = b1[i];
        Vts[(c0 + 16 + i) * 72 + row] = b2[i];
        Vts[(c0 + 24 + i) * 72 + row] = b3[i];
      }
      wait_async0();
    }
    __syncthreads();

    // scores S = Q * K^T for 4 blocks of 16 keys
    v8f sc[4];
#pragma unroll
    for (int j = 0; j < 4; ++j) {
      sc[j] = vzero8();
#pragma unroll
      for (int f = 0; f < 4; ++f) {
        v16bf kf = ldfrag(&Ksm[(j * 16 + rsel) * 136 + f * 32 + hsel * 8]);
        sc[j] = wmma_bf16(qf[f], kf, sc[j]);
      }
    }

    // scale + causal mask + row max
    float rmax[8];
#pragma unroll
    for (int r = 0; r < 8; ++r) rmax[r] = -1e30f;
#pragma unroll
    for (int j = 0; j < 4; ++j) {
      const int tg = kbase + j * 16 + rsel;
#pragma unroll
      for (int r = 0; r < 8; ++r) {
        const int rg = q0 + r + 8 * hsel;
        const float v = (tg <= rg) ? sc[j][r] * scale : -1e9f;
        sc[j][r] = v;
        rmax[r] = fmaxf(rmax[r], v);
      }
    }
#pragma unroll
    for (int r = 0; r < 8; ++r)
#pragma unroll
      for (int m = 1; m < 16; m <<= 1)
        rmax[r] = fmaxf(rmax[r], __shfl_xor(rmax[r], m, 32));

    float alpha[8], rsum[8];
#pragma unroll
    for (int r = 0; r < 8; ++r) {
      const float mn = fmaxf(m_s[r], rmax[r]);
      alpha[r] = __expf(m_s[r] - mn);
      m_s[r] = mn;
      rsum[r] = 0.0f;
    }
#pragma unroll
    for (int j = 0; j < 4; ++j)
#pragma unroll
      for (int r = 0; r < 8; ++r) {
        const float p = __expf(sc[j][r] - m_s[r]);
        sc[j][r] = p;
        rsum[r] += p;
      }
#pragma unroll
    for (int r = 0; r < 8; ++r)
#pragma unroll
      for (int m = 1; m < 16; m <<= 1) rsum[r] += __shfl_xor(rsum[r], m, 32);
#pragma unroll
    for (int r = 0; r < 8; ++r) l_s[r] = l_s[r] * alpha[r] + rsum[r];
#pragma unroll
    for (int oc = 0; oc < 8; ++oc)
#pragma unroll
      for (int r = 0; r < 8; ++r) oacc[oc][r] *= alpha[r];

    // route P through per-wave LDS (C layout -> A layout)
    unsigned short* pw = &Psm[w * 16 * 72];
#pragma unroll
    for (int j = 0; j < 4; ++j)
#pragma unroll
      for (int r = 0; r < 8; ++r)
        pw[(r + 8 * hsel) * 72 + j * 16 + rsel] = f2bf(sc[j][r]);
    asm volatile("s_wait_dscnt 0x0" ::: "memory");  // intra-wave LDS RAW

    v16bf pf[2];
#pragma unroll
    for (int f = 0; f < 2; ++f)
      pf[f] = ldfragA(&pw[rsel * 72 + f * 32 + hsel * 8]);
#pragma unroll
    for (int oc = 0; oc < 8; ++oc)
#pragma unroll
      for (int f = 0; f < 2; ++f) {
        v16bf vf = ldfrag(&Vts[(oc * 16 + rsel) * 72 + f * 32 + hsel * 8]);
        oacc[oc] = wmma_bf16(pf[f], vf, oacc[oc]);
      }
  }

  // epilogue: normalize and store O as bf16 [B,S,D]
#pragma unroll
  for (int oc = 0; oc < 8; ++oc)
#pragma unroll
    for (int r = 0; r < 8; ++r) {
      const int rg = q0 + r + 8 * hsel;         // token s
      const int dg = h * 128 + oc * 16 + rsel;  // column in D
      const float v = oacc[oc][r] / l_s[r];
      O[((size_t)b * 2048 + rg) * 2048 + dg] = f2bf(v);
    }
}

// ---------------- host orchestration ----------------
extern "C" void kernel_launch(void* const* d_in, const int* in_sizes, int n_in,
                              void* d_out, int out_size, void* d_ws, size_t ws_size,
                              hipStream_t stream) {
  (void)in_sizes; (void)n_in; (void)out_size; (void)ws_size;
  const float* x = (const float*)d_in[0];
  const float* wq = (const float*)d_in[1];
  const float* wk = (const float*)d_in[2];
  const float* wv = (const float*)d_in[3];
  const float* wo = (const float*)d_in[4];
  const float* fc = (const float*)d_in[5];
  const float* fs = (const float*)d_in[6];
  // d_in[7] (mask) unused: causality applied analytically.

  char* ws = (char*)d_ws;
  const size_t WB = (size_t)2048 * 2048 * 2;      // 8 MiB  bf16 weight
  const size_t TB = (size_t)2 * 2048 * 2048 * 2;  // 16 MiB bf16 activation
  unsigned short* wqb = (unsigned short*)(ws);
  unsigned short* wkb = (unsigned short*)(ws + WB);
  unsigned short* wvb = (unsigned short*)(ws + 2 * WB);
  unsigned short* wob = (unsigned short*)(ws + 3 * WB);
  unsigned short* xb = (unsigned short*)(ws + 4 * WB);
  unsigned short* qh = (unsigned short*)(ws + 4 * WB + TB);
  unsigned short* kh = (unsigned short*)(ws + 4 * WB + 2 * TB);
  unsigned short* vh = (unsigned short*)(ws + 4 * WB + 3 * TB);
  unsigned short* Ob = (unsigned short*)(ws + 4 * WB + 4 * TB);
  // total workspace: 4*8 + 5*16 = 112 MiB

  const int NW = 2048 * 2048;
  const int NX = 2 * 2048 * 2048;
  cvt_f32_bf16<<<NW / 256, 256, 0, stream>>>(wq, wqb, NW);
  cvt_f32_bf16<<<NW / 256, 256, 0, stream>>>(wk, wkb, NW);
  cvt_f32_bf16<<<NW / 256, 256, 0, stream>>>(wv, wvb, NW);
  cvt_f32_bf16<<<NW / 256, 256, 0, stream>>>(wo, wob, NW);
  cvt_f32_bf16<<<NX / 256, 256, 0, stream>>>(x, xb, NX);

  dim3 gg(16, 32);  // N/128, M/128  (M = B*S = 4096, N = D = 2048)
  gemm_nt<true, true><<<gg, 256, 0, stream>>>(xb, wqb, qh, 4096, 2048, 2048);
  gemm_nt<true, true><<<gg, 256, 0, stream>>>(xb, wkb, kh, 4096, 2048, 2048);
  gemm_nt<true, true><<<gg, 256, 0, stream>>>(xb, wvb, vh, 4096, 2048, 2048);

  const int NP = 2 * 16 * 2048 * 64;  // rotation pairs
  rope_inplace<<<NP / 256, 256, 0, stream>>>(qh, fc, fs);
  rope_inplace<<<NP / 256, 256, 0, stream>>>(kh, fc, fs);

  flash_attn<<<dim3(16, 16, 2), 256, 0, stream>>>(qh, kh, vh, Ob);

  gemm_nt<false, false><<<gg, 256, 0, stream>>>(Ob, wob, d_out, 4096, 2048, 2048);
}
